// AttentionHead_32358283608297
// MI455X (gfx1250) — matrix operand order repaired and checked by reading, not compiled, not run on hardware
//
#include <hip/hip_runtime.h>
#include <cstdint>

typedef __attribute__((ext_vector_type(16))) _Float16 v16h;
typedef __attribute__((ext_vector_type(8)))  float    v8f;

constexpr int S_LEN = 2048;
constexpr int E_DIM = 1024;
constexpr int H_DIM = 64;
constexpr int BATCH = 8;

union Frag16 {
  v16h v;
  _Float16 h[16];
  uint4 q[2];
};

__device__ __forceinline__ v8f wmma_f16(v16h a, v16h b, v8f c) {
  // v_wmma_f32_16x16x32_f16: D = A(16x32 f16) * B(32x16 f16) + C(16x16 f32)
  return __builtin_amdgcn_wmma_f32_16x16x32_f16(
      /*neg_a=*/false, a, /*neg_b=*/false, b,
      /*c_mod=*/(short)0, c, /*reuse_a=*/false, /*reuse_b=*/false);
}

// Async global->LDS copy, 16B per lane (GLOBAL_LOAD_ASYNC_TO_LDS_B128).
// VDST = LDS byte address (wave-relative, low 32 bits of flat shared ptr),
// VADDR = 64-bit global address, GV mode (saddr = off). Tracked by ASYNCcnt.
__device__ __forceinline__ void async_copy_b128(unsigned lds_off,
                                                const void* gaddr) {
  asm volatile("global_load_async_to_lds_b128 %0, %1, off"
               :
               : "v"(lds_off), "v"(gaddr)
               : "memory");
}

__device__ __forceinline__ void wait_async0() {
  asm volatile("s_wait_asynccnt 0x0" ::: "memory");
}

__device__ __forceinline__ unsigned lds_off_of(const void* p) {
  return (unsigned)(uintptr_t)p;  // flat shared ptr: LDS offset in [31:0]
}

// ---------------------------------------------------------------------------
// Kernel 1: cast projection weights fp32 -> fp16 (kept in [h][e] row-major)
// ---------------------------------------------------------------------------
__global__ void cast_weights(const float* __restrict__ Wq,
                             const float* __restrict__ Wk,
                             const float* __restrict__ Wv,
                             _Float16* __restrict__ Wh) {
  int i = blockIdx.x * blockDim.x + threadIdx.x;
  if (i < H_DIM * E_DIM) {
    Wh[i]                     = (_Float16)Wq[i];
    Wh[H_DIM * E_DIM + i]     = (_Float16)Wk[i];
    Wh[2 * H_DIM * E_DIM + i] = (_Float16)Wv[i];
  }
}

// ---------------------------------------------------------------------------
// Kernel 2: QKV projection. One wave computes a 16-row tile of Q,K,V (64 cols
// each) with WMMA over E=1024 in K-chunks of 32. Q,K stored [row][h] f16;
// V stored transposed [b][h][s] f16 so the PV B-fragments load contiguously.
// ---------------------------------------------------------------------------
__global__ __launch_bounds__(128) void qkv_project(
    const float* __restrict__ x, const _Float16* __restrict__ Wh,
    const float* __restrict__ bq, const float* __restrict__ bk,
    const float* __restrict__ bv,
    _Float16* __restrict__ Qh, _Float16* __restrict__ Kh,
    _Float16* __restrict__ Vt) {
  const int lane = threadIdx.x & 31;
  const int wave = threadIdx.x >> 5;
  const int tile = blockIdx.x * 4 + wave;      // 16-row tile of flattened B*S
  const int row0 = tile * 16;
  const int m16  = lane & 15;
  const int hs   = lane >> 4;

  v8f acc[3][4];
  const v8f vzero = {};
#pragma unroll
  for (int m = 0; m < 3; ++m)
#pragma unroll
    for (int n = 0; n < 4; ++n) acc[m][n] = vzero;

  for (int koff = 0; koff < E_DIM; koff += 32) {
    // A fragment of x (16x32, f16): lane holds row m16; K pattern per ISA:
    // elem i<8  -> k = hs*8 + i ; elem i>=8 -> k = 16 + hs*8 + (i-8)
    const float* xr = x + (size_t)(row0 + m16) * E_DIM + koff + hs * 8;
    __builtin_prefetch(xr + 32, 0, 1);           // global_prefetch next chunk
    const float4 f0 = *(const float4*)(xr + 0);
    const float4 f1 = *(const float4*)(xr + 4);
    const float4 f2 = *(const float4*)(xr + 16);
    const float4 f3 = *(const float4*)(xr + 20);
    const float xa[16] = {f0.x, f0.y, f0.z, f0.w, f1.x, f1.y, f1.z, f1.w,
                          f2.x, f2.y, f2.z, f2.w, f3.x, f3.y, f3.z, f3.w};
    Frag16 a;
#pragma unroll
    for (int i = 0; i < 16; ++i) a.h[i] = (_Float16)xa[i];

#pragma unroll
    for (int mat = 0; mat < 3; ++mat) {
#pragma unroll
      for (int nt = 0; nt < 4; ++nt) {
        // B fragment of W^T: n = nt*16 + m16 (output col), k = hs*8 + i / 16 + hs*8 + (i-8)
        const _Float16* wr = Wh + (size_t)mat * (H_DIM * E_DIM) +
                             (size_t)(nt * 16 + m16) * E_DIM + koff + hs * 8;
        Frag16 bf;
        bf.q[0] = *(const uint4*)(wr + 0);
        bf.q[1] = *(const uint4*)(wr + 16);
        acc[mat][nt] = wmma_f16(a.v, bf.v, acc[mat][nt]);
      }
    }
  }

  const float* bias[3] = {bq, bk, bv};
#pragma unroll
  for (int mat = 0; mat < 3; ++mat) {
#pragma unroll
    for (int nt = 0; nt < 4; ++nt) {
      const int n = nt * 16 + m16;
      const float bb = bias[mat][n];
#pragma unroll
      for (int r = 0; r < 8; ++r) {
        const int row = row0 + r + 8 * hs;       // D layout: m = r + 8*(lane/16)
        const _Float16 val = (_Float16)(acc[mat][nt][r] + bb);
        if (mat == 0) {
          Qh[(size_t)row * H_DIM + n] = val;
        } else if (mat == 1) {
          Kh[(size_t)row * H_DIM + n] = val;
        } else {
          const int bi = row >> 11;              // row / 2048
          const int si = row & (S_LEN - 1);
          Vt[((size_t)bi * H_DIM + n) * S_LEN + si] = val;
        }
      }
    }
  }
}

// ---------------------------------------------------------------------------
// Kernel 3: causal flash attention. One wave owns (batch, 16 query rows) and
// streams keys in chunks of 32 with online softmax. K/V tiles for the next
// chunk are double-buffered into LDS with async global->LDS copies (ASYNCcnt)
// so the async unit overlaps the current chunk's WMMAs. fp32 accumulation.
// ---------------------------------------------------------------------------
__global__ __launch_bounds__(128) void flash_attn(
    const _Float16* __restrict__ Qh, const _Float16* __restrict__ Kh,
    const _Float16* __restrict__ Vt, float* __restrict__ out) {
  __shared__ _Float16 kbuf[4][2][32 * H_DIM];    // 32 keys x 64 h  (4KB/tile)
  __shared__ _Float16 vbuf[4][2][H_DIM * 32];    // 64 h  x 32 keys (4KB/tile)
  __shared__ float pstage[4][16][32];            // per-wave P tile restage
  const int lane  = threadIdx.x & 31;
  const int wave  = threadIdx.x >> 5;
  const int wid   = blockIdx.x * 4 + wave;       // 0..1023
  const int b     = wid >> 7;                    // 128 q-tiles per batch
  const int qbase = (wid & 127) * 16;
  const int m16   = lane & 15;
  const int hs    = lane >> 4;

  const _Float16* Kb = Kh + (size_t)b * S_LEN * H_DIM;
  const _Float16* Vb = Vt + (size_t)b * H_DIM * S_LEN;

  // --- async prefetch of one 32-key K/V tile pair into LDS buffer `buf` ---
  auto prefetch_kv = [&](int kb, int buf) {
    // K tile: 32 consecutive rows of 128B -> one contiguous 4KB block
    {
      const char* src = (const char*)(Kb + (size_t)kb * H_DIM);
      const unsigned dst = lds_off_of(&kbuf[wave][buf][0]);
#pragma unroll
      for (int j = 0; j < 8; ++j) {
        const unsigned off = (unsigned)(j * 32 + lane) * 16u;
        async_copy_b128(dst + off, src + off);
      }
    }
    // V^T tile: 64 rows x 64B (row stride 4096B in global, 64B in LDS)
    {
      const int r   = lane >> 2;                 // 0..7
      const int cst = (lane & 3) * 16;           // 16B chunk within row
      const unsigned dst = lds_off_of(&vbuf[wave][buf][0]);
#pragma unroll
      for (int j = 0; j < 8; ++j) {
        const int row = j * 8 + r;
        const char* src =
            (const char*)(Vb + (size_t)row * S_LEN + kb) + cst;
        async_copy_b128(dst + (unsigned)(row * 64 + cst), src);
      }
    }
  };

  // Q A-fragments for head-dim chunks [0,32) and [32,64)
  const _Float16* qr = Qh + (size_t)(b * S_LEN + qbase + m16) * H_DIM;
  Frag16 aq0, aq1;
  aq0.q[0] = *(const uint4*)(qr + hs * 8);
  aq0.q[1] = *(const uint4*)(qr + hs * 8 + 16);
  aq1.q[0] = *(const uint4*)(qr + 32 + hs * 8);
  aq1.q[1] = *(const uint4*)(qr + 32 + hs * 8 + 16);

  const v8f vzero = {};
  v8f o[4];
  float rm[8], rl[8];
#pragma unroll
  for (int n = 0; n < 4; ++n) o[n] = vzero;
#pragma unroll
  for (int r = 0; r < 8; ++r) { rm[r] = -__builtin_inff(); rl[r] = 0.f; }

  const int nch = (qbase + 16 + 31) >> 5;        // causal: keys < qbase+16
  int cur = 0;
  prefetch_kv(0, 0);

  for (int c = 0; c < nch; ++c) {
    const int kb = c * 32;
    wait_async0();                               // current tiles resident
    if (c + 1 < nch) prefetch_kv((c + 1) * 32, cur ^ 1);
    const _Float16* kt = &kbuf[wave][cur][0];
    const _Float16* vt = &vbuf[wave][cur][0];

    // --- scores S = Q * K^T for 32 keys (two 16-col tiles), from LDS ---
    v8f s[2];
#pragma unroll
    for (int t = 0; t < 2; ++t) {
      const _Float16* kr = kt + (t * 16 + m16) * H_DIM;
      Frag16 b0, b1;
      b0.q[0] = *(const uint4*)(kr + hs * 8);
      b0.q[1] = *(const uint4*)(kr + hs * 8 + 16);
      b1.q[0] = *(const uint4*)(kr + 32 + hs * 8);
      b1.q[1] = *(const uint4*)(kr + 32 + hs * 8 + 16);
      v8f st = vzero;
      st = wmma_f16(aq0.v, b0.v, st);
      st = wmma_f16(aq1.v, b1.v, st);
      s[t] = st;
    }

    // --- causal mask (before scale, matching reference), then scale ---
    if (kb + 31 > qbase) {
#pragma unroll
      for (int r = 0; r < 8; ++r) {
        const int qi = qbase + r + 8 * hs;
        if (kb + m16 > qi)      s[0][r] = -__builtin_inff();
        if (kb + 16 + m16 > qi) s[1][r] = -__builtin_inff();
      }
    }
    float mx[8];
#pragma unroll
    for (int r = 0; r < 8; ++r) {
      s[0][r] *= 0.125f;                          // 1/sqrt(64)
      s[1][r] *= 0.125f;
      mx[r] = fmaxf(s[0][r], s[1][r]);
    }
#pragma unroll
    for (int off = 1; off < 16; off <<= 1)
#pragma unroll
      for (int r = 0; r < 8; ++r)
        mx[r] = fmaxf(mx[r], __shfl_xor(mx[r], off, 32));

    // --- online softmax update ---
    float sum[8];
#pragma unroll
    for (int r = 0; r < 8; ++r) {
      const float mn   = fmaxf(rm[r], mx[r]);
      const float corr = __expf(rm[r] - mn);
      rm[r] = mn;
      s[0][r] = __expf(s[0][r] - mn);
      s[1][r] = __expf(s[1][r] - mn);
      rl[r] *= corr;
      sum[r] = s[0][r] + s[1][r];
#pragma unroll
      for (int n = 0; n < 4; ++n) o[n][r] *= corr;
    }
#pragma unroll
    for (int off = 1; off < 16; off <<= 1)
#pragma unroll
      for (int r = 0; r < 8; ++r) sum[r] += __shfl_xor(sum[r], off, 32);
#pragma unroll
    for (int r = 0; r < 8; ++r) rl[r] += sum[r];

    // --- repack P (D layout f32) into an A-fragment via LDS ---
#pragma unroll
    for (int r = 0; r < 8; ++r) {
      pstage[wave][r + 8 * hs][m16]      = s[0][r];
      pstage[wave][r + 8 * hs][16 + m16] = s[1][r];
    }
    __threadfence_block();                        // wait ds writes (same wave)
    Frag16 ap;
#pragma unroll
    for (int i = 0; i < 8; ++i)
      ap.h[i] = (_Float16)pstage[wave][m16][hs * 16 + i];
#pragma unroll
    for (int i = 0; i < 8; ++i)
      ap.h[8 + i] = (_Float16)pstage[wave][m16][hs * 16 + 8 + i];

    // --- O += P * V  (V^T tile: contiguous key pairs per B-fragment) ---
#pragma unroll
    for (int nt = 0; nt < 4; ++nt) {
      const _Float16* vr = vt + (nt * 16 + m16) * 32 + hs * 16;
      Frag16 bf;
      bf.q[0] = *(const uint4*)(vr + 0);
      bf.q[1] = *(const uint4*)(vr + 8);
      o[nt] = wmma_f16(ap.v, bf.v, o[nt]);
    }
    cur ^= 1;
  }

  // --- normalize and store (fp32) ---
  float* orow = out + (size_t)(b * S_LEN + qbase) * H_DIM;
#pragma unroll
  for (int nt = 0; nt < 4; ++nt)
#pragma unroll
    for (int r = 0; r < 8; ++r) {
      const int m = r + 8 * hs;
      orow[(size_t)m * H_DIM + nt * 16 + m16] = o[nt][r] / rl[r];
    }
}

// ---------------------------------------------------------------------------
extern "C" void kernel_launch(void* const* d_in, const int* in_sizes, int n_in,
                              void* d_out, int out_size, void* d_ws,
                              size_t ws_size, hipStream_t stream) {
  const float* x  = (const float*)d_in[0];
  const float* Wq = (const float*)d_in[1];
  const float* bq = (const float*)d_in[2];
  const float* Wk = (const float*)d_in[3];
  const float* bk = (const float*)d_in[4];
  const float* Wv = (const float*)d_in[5];
  const float* bv = (const float*)d_in[6];
  float* out = (float*)d_out;

  // Workspace layout (halves): Wh[3*64*1024] | Qh | Kh | Vt (each B*S*H)
  _Float16* Wh = (_Float16*)d_ws;
  _Float16* Qh = Wh + (size_t)3 * H_DIM * E_DIM;
  _Float16* Kh = Qh + (size_t)BATCH * S_LEN * H_DIM;
  _Float16* Vt = Kh + (size_t)BATCH * S_LEN * H_DIM;

  cast_weights<<<(H_DIM * E_DIM + 255) / 256, 256, 0, stream>>>(Wq, Wk, Wv, Wh);
  qkv_project<<<(BATCH * S_LEN) / (16 * 4), 128, 0, stream>>>(
      x, Wh, bq, bk, bv, Qh, Kh, Vt);
  flash_attn<<<(BATCH * (S_LEN / 16)) / 4, 128, 0, stream>>>(Qh, Kh, Vt, out);
}
